// SparseConvolution_4415226380611
// MI455X (gfx1250) — compile-verified, matrix operand order checked by reading, not run on hardware
//
#include <hip/hip_runtime.h>

// ---------------------------------------------------------------------------
// Types for CDNA5 WMMA (wave32, 16x16x32 bf16 -> f32)
// ---------------------------------------------------------------------------
typedef __bf16 v16bf __attribute__((ext_vector_type(16)));
typedef __bf16 v8bf  __attribute__((ext_vector_type(8)));
typedef float  v8f   __attribute__((ext_vector_type(8)));

static __device__ __forceinline__ unsigned short f2bfu(float f) {
    __bf16 h = (__bf16)f;
    return __builtin_bit_cast(unsigned short, h);
}

// ---------------------------------------------------------------------------
// Gathered convolution, fragment-swizzled weights, zero-padded activations.
//   out[Mout, CO] = sum_k gather(feat, nbr[k]) @ W[k]
// feat: bf16 [Min, CIp] (CIp = ceil32(CI), pad columns zero)
// wswz: bf16 in B-fragment order [ntaps][CIp/32][CO/16][lane=32][elem=16]
// nbr : int32 [ntaps, Mout] or nullptr (identity). Negative index -> zero row.
// One wave computes a 16 x (16*NT) output tile. All hot-loop loads are
// unconditional b128s; the only WMMA inputs are register fragments.
// ---------------------------------------------------------------------------
template <int NT>
__global__ __launch_bounds__(32) void k_gconv(
    const unsigned short* __restrict__ feat,
    const unsigned short* __restrict__ wswz,
    const int* __restrict__ nbr,
    const unsigned short* __restrict__ zrow,
    float* __restrict__ out,
    int Mout, int CIp, int CO, int ntaps)
{
    const int lane = threadIdx.x;        // 0..31
    const int m16  = lane & 15;
    const int hi   = lane >> 4;          // 0 or 1
    const int row  = blockIdx.x * 16 + m16;
    const int ntilesN = CO >> 4;
    const int nchunks = CIp >> 5;
    const int tile0   = blockIdx.y * NT;

    v8f acc[NT];
#pragma unroll
    for (int t = 0; t < NT; ++t) acc[t] = (v8f){};

    for (int k = 0; k < ntaps; ++k) {
        int src = -1;
        if (row < Mout) src = nbr ? nbr[(size_t)k * Mout + row] : row;
        const unsigned short* arow = (src >= 0) ? feat + (size_t)src * CIp : zrow;
        const unsigned short* wk   = wswz + (size_t)k * nchunks * ntilesN * 512;

        for (int c = 0; c < nchunks; ++c) {
            // A fragment (ISA 7.12.2): elems 0-7 -> K = 32c + 8*hi + 0..7
            //                          elems 8-15 -> K = 32c + 16 + 8*hi + 0..7
            v8bf alo = *(const v8bf*)(arow + c * 32 + hi * 8);
            v8bf ahi = *(const v8bf*)(arow + c * 32 + 16 + hi * 8);
            v16bf a = __builtin_shufflevector(
                alo, ahi, 0, 1, 2, 3, 4, 5, 6, 7, 8, 9, 10, 11, 12, 13, 14, 15);

            const unsigned short* wc =
                wk + (((size_t)c * ntilesN + tile0) * 32 + lane) * 16;
#pragma unroll
            for (int t = 0; t < NT; ++t) {
                v16bf b = *(const v16bf*)(wc + (size_t)t * 512);
                acc[t] = __builtin_amdgcn_wmma_f32_16x16x32_bf16(
                    false, a, false, b, (short)0, acc[t], false, false);
            }
        }
    }

    // D layout: VGPR v of lane l -> row (v + hi*8), col (l&15)
#pragma unroll
    for (int t = 0; t < NT; ++t) {
#pragma unroll
        for (int v = 0; v < 8; ++v) {
            int r = blockIdx.x * 16 + v + hi * 8;
            if (r < Mout)
                out[(size_t)r * CO + (tile0 + t) * 16 + m16] = acc[t][v];
        }
    }
}

// ---------------------------------------------------------------------------
// Weight swizzle: W f32 [ntaps, CI, CO] -> bf16 B-fragment order
// [ntaps][CIp/32][CO/16][lane][elem], zero-padded K.
// ---------------------------------------------------------------------------
__global__ __launch_bounds__(256) void k_swz(
    const float* __restrict__ W, unsigned short* __restrict__ out,
    int ntaps, int CI, int CIp, int CO, size_t n)
{
    size_t i = (size_t)blockIdx.x * 256 + threadIdx.x;
    if (i >= n) return;
    int e    = (int)(i & 15);
    int lane = (int)((i >> 4) & 31);
    size_t rest = i >> 9;
    int ntilesN = CO >> 4, nchunks = CIp >> 5;
    int tn = (int)(rest % ntilesN); rest /= ntilesN;
    int c  = (int)(rest % nchunks);
    int k  = (int)(rest / nchunks);
    int hi = lane >> 4;
    int kk = c * 32 + ((e < 8) ? e : e + 8) + hi * 8;
    int col = tn * 16 + (lane & 15);
    float v = (kk < CI) ? W[((size_t)k * CI + kk) * CO + col] : 0.f;
    out[i] = f2bfu(v);
}

// ---------------------------------------------------------------------------
// BatchNorm statistics (biased variance, matching jnp.var)
// ---------------------------------------------------------------------------
__global__ __launch_bounds__(256) void k_bn_stats(
    const float* __restrict__ x, int M, int C,
    float* __restrict__ mean, float* __restrict__ inv)
{
    const int c = blockIdx.x;
    float s = 0.f, s2 = 0.f;
    for (int r = threadIdx.x; r < M; r += 256) {
        float v = x[(size_t)r * C + c];
        s += v; s2 += v * v;
    }
    __shared__ float sh[256], sh2[256];
    sh[threadIdx.x] = s; sh2[threadIdx.x] = s2;
    __syncthreads();
    for (int o = 128; o > 0; o >>= 1) {
        if (threadIdx.x < o) {
            sh[threadIdx.x]  += sh[threadIdx.x + o];
            sh2[threadIdx.x] += sh2[threadIdx.x + o];
        }
        __syncthreads();
    }
    if (threadIdx.x == 0) {
        float m   = sh[0] / (float)M;
        float var = sh2[0] / (float)M - m * m;
        mean[c] = m;
        inv[c]  = rsqrtf(var + 1e-5f);
    }
}

// mode: 0 = bn, 1 = relu(bn), 2 = relu(bn + add), 3 = add + relu(bn) (accum)
// outf is dense [M,C]; outbf is padded [M,Cp] with zeroed pad columns.
__global__ __launch_bounds__(256) void k_bn_apply(
    const float* __restrict__ x,
    const float* __restrict__ mean, const float* __restrict__ inv,
    const float* __restrict__ g, const float* __restrict__ b,
    const float* __restrict__ add,
    float* __restrict__ outf, unsigned short* __restrict__ outbf,
    int M, int C, int Cp, int mode)
{
    size_t i = (size_t)blockIdx.x * 256 + threadIdx.x;
    if (i >= (size_t)M * Cp) return;
    int c = (int)(i % Cp);
    int r = (int)(i / Cp);
    if (c >= C) { if (outbf) outbf[i] = 0; return; }
    size_t j = (size_t)r * C + c;
    float y = (x[j] - mean[c]) * inv[c] * g[c] + b[c];
    if (mode == 2) y += add[j];
    if (mode >= 1) y = fmaxf(y, 0.f);
    if (mode == 3) y += add[j];
    if (outf)  outf[j]  = y;
    if (outbf) outbf[i] = f2bfu(y);
}

// ---------------------------------------------------------------------------
// Elementwise helpers
// ---------------------------------------------------------------------------
__global__ __launch_bounds__(256) void k_zero_f32(float* __restrict__ p, size_t n) {
    size_t i = (size_t)blockIdx.x * 256 + threadIdx.x;
    if (i < n) p[i] = 0.f;
}

__global__ __launch_bounds__(256) void k_scatter_add(
    const float* __restrict__ pts, const int* __restrict__ p2v,
    float* __restrict__ acc, int Np, int C)
{
    size_t i = (size_t)blockIdx.x * 256 + threadIdx.x;
    if (i >= (size_t)Np * C) return;
    int p = (int)(i / C), c = (int)(i % C);
    atomicAdd(&acc[(size_t)p2v[p] * C + c], pts[i]);
}

// acc (dense [M,C]) /= counts; bf16 copy with padded stride Cp
__global__ __launch_bounds__(256) void k_div_counts(
    float* __restrict__ acc, const float* __restrict__ counts,
    unsigned short* __restrict__ outbf, int M, int C, int Cp)
{
    size_t i = (size_t)blockIdx.x * 256 + threadIdx.x;
    if (i >= (size_t)M * Cp) return;
    int c = (int)(i % Cp);
    int r = (int)(i / Cp);
    if (c >= C) { if (outbf) outbf[i] = 0; return; }
    size_t j = (size_t)r * C + c;
    float v = acc[j] / counts[r];
    acc[j] = v;
    if (outbf) outbf[i] = f2bfu(v);
}

// devox: out[p,c] = sum_k w8[p,k] * fv[idx8[p,k], c]  (idx<0 -> 0)
// C here is always a multiple of 32, so bf16 output needs no padding.
__global__ __launch_bounds__(256) void k_devox(
    const float* __restrict__ fv, const int* __restrict__ idx8,
    const float* __restrict__ w8,
    float* __restrict__ out, unsigned short* __restrict__ outbf,
    int Np, int C)
{
    size_t i = (size_t)blockIdx.x * 256 + threadIdx.x;
    if (i >= (size_t)Np * C) return;
    int p = (int)(i / C), c = (int)(i % C);
    float s = 0.f;
#pragma unroll
    for (int k = 0; k < 8; ++k) {
        int id = idx8[(size_t)p * 8 + k];
        if (id >= 0) s += w8[(size_t)p * 8 + k] * fv[(size_t)id * C + c];
    }
    out[i] = s;
    if (outbf) outbf[i] = f2bfu(s);
}

// copy bf16 [M x Cs] (row stride srcCp) into columns [off, off+Cs) of
// bf16 [M x *] with row stride dstCp
__global__ __launch_bounds__(256) void k_copy_cols(
    const unsigned short* __restrict__ src, unsigned short* __restrict__ dst,
    int M, int Cs, int srcCp, int dstCp, int off)
{
    size_t i = (size_t)blockIdx.x * 256 + threadIdx.x;
    if (i >= (size_t)M * Cs) return;
    int r = (int)(i / Cs), c = (int)(i % Cs);
    dst[(size_t)r * dstCp + off + c] = src[(size_t)r * srcCp + c];
}

// zero columns [from, to) of bf16 [M x stride]
__global__ __launch_bounds__(256) void k_pad_cols(
    unsigned short* __restrict__ dst, int M, int stride, int from, int to)
{
    int w = to - from;
    size_t i = (size_t)blockIdx.x * 256 + threadIdx.x;
    if (i >= (size_t)M * w) return;
    int r = (int)(i / w), c = (int)(i % w);
    dst[(size_t)r * stride + from + c] = 0;
}

// ---------------------------------------------------------------------------
// Host orchestration
// ---------------------------------------------------------------------------
extern "C" void kernel_launch(void* const* d_in, const int* in_sizes, int n_in,
                              void* d_out, int out_size, void* d_ws, size_t ws_size,
                              hipStream_t stream)
{
    (void)out_size; (void)ws_size;
    if (n_in < 145) return;

    const int CS[9] = {16, 32, 64, 128, 256, 256, 128, 64, 32};

    // ---- input index map (setup_inputs dict order) ----
    int pi = 1;
    int iStemW = pi++, iStemG = pi++, iStemB = pi++;
    struct PS { int cw, cg, cb, w1, g1, b1, w2, g2, b2, wd, gd, bd; };
    PS S[4], U[4];
    for (int i = 0; i < 4; ++i) { PS& s = S[i];
        s.cw=pi++; s.cg=pi++; s.cb=pi++; s.w1=pi++; s.g1=pi++; s.b1=pi++;
        s.w2=pi++; s.g2=pi++; s.b2=pi++; s.wd=pi++; s.gd=pi++; s.bd=pi++; }
    for (int i = 0; i < 4; ++i) { PS& s = U[i];
        s.cw=pi++; s.cg=pi++; s.cb=pi++; s.w1=pi++; s.g1=pi++; s.b1=pi++;
        s.w2=pi++; s.g2=pi++; s.b2=pi++; s.wd=pi++; s.gd=pi++; s.bd=pi++; }
    int iTW[5], iTg[5], iTb[5];
    for (int i = 0; i < 5; ++i) { iTW[i]=pi++; pi++; /* bias cancels in BN */ iTg[i]=pi++; iTb[i]=pi++; }
    int iP2V = pi++, iCnt = pi++;
    int iN27[5];  for (int l = 0; l < 5; ++l) iN27[l] = pi++;
    int iDown[5]; for (int l = 1; l < 5; ++l) iDown[l] = pi++;
    int iDec[5];  for (int l = 1; l < 5; ++l) iDec[l] = pi++;
    int iVI[5], iVW[5];
    for (int l = 0; l < 5; ++l) { iVI[l] = pi++; iVW[l] = pi++; }

    const int Npts = in_sizes[0] / 3;
    int M[5];
    for (int l = 0; l < 5; ++l) M[l] = in_sizes[iN27[l]] / 27;
    const int M0 = M[0];

    auto P = [&](int i) { return (const float*)d_in[i]; };
    auto I = [&](int i) { return (const int*)d_in[i]; };
    auto pad32 = [](int c) { return (c + 31) & ~31; };
    auto mx = [](size_t a, size_t b) { return a > b ? a : b; };

    // ---- bump allocator over d_ws ----
    char* base = (char*)d_ws;
    size_t off = 0;
    auto alloc = [&](size_t bytes) -> void* {
        void* r = base + off;
        off = (off + bytes + 255) & ~(size_t)255;
        return r;
    };

    // scratch sizing (f32 temps dense; bf16 buffers padded)
    size_t tmax = (size_t)M0 * 16;
    for (int s = 0; s < 4; ++s) {
        tmax = mx(tmax, (size_t)M[s + 1] * CS[s + 1]);
        tmax = mx(tmax, (size_t)M[s + 1] * CS[s]);
    }
    size_t catmax = (size_t)M0 * 32;          // down s=0 padded cat
    for (int s = 0; s < 4; ++s)
        catmax = mx(catmax, (size_t)M[s + 1] * pad32(CS[s]));
    for (int u = 0; u < 4; ++u) {
        int lev = 3 - u, co = CS[5 + u], cc = co + CS[3 - u];
        tmax   = mx(tmax,   (size_t)M[lev] * co);
        catmax = mx(catmax, (size_t)M[lev] * pad32(cc));
    }
    tmax = mx(tmax, (size_t)Npts * 32);
    catmax = mx(catmax, tmax);

    // ---- launch helpers ----
    auto nblk = [](size_t n) { return (unsigned)((n + 255) / 256); };

    unsigned short* zrow = (unsigned short*)alloc(384 * 2);
    k_zero_f32<<<1, 256, 0, stream>>>((float*)zrow, 192);

    auto conv = [&](const void* f, const void* w, const int* nbr, float* o,
                    int Mo, int CIp_, int CO_, int nt) {
        unsigned gx = (unsigned)((Mo + 15) / 16);
        if (CO_ == 16)
            k_gconv<1><<<dim3(gx, 1), 32, 0, stream>>>(
                (const unsigned short*)f, (const unsigned short*)w, nbr, zrow,
                o, Mo, CIp_, CO_, nt);
        else if (CO_ == 32)
            k_gconv<2><<<dim3(gx, 1), 32, 0, stream>>>(
                (const unsigned short*)f, (const unsigned short*)w, nbr, zrow,
                o, Mo, CIp_, CO_, nt);
        else
            k_gconv<4><<<dim3(gx, (unsigned)(CO_ / 64)), 32, 0, stream>>>(
                (const unsigned short*)f, (const unsigned short*)w, nbr, zrow,
                o, Mo, CIp_, CO_, nt);
    };

    float* statm = (float*)alloc(512 * 4);
    float* stati = (float*)alloc(512 * 4);
    auto bn = [&](const float* x, int Mo, int C, int Cp, const float* g,
                  const float* b, int mode, const float* add,
                  float* outf, void* outbf) {
        k_bn_stats<<<dim3((unsigned)C), 256, 0, stream>>>(x, Mo, C, statm, stati);
        k_bn_apply<<<nblk((size_t)Mo * Cp), 256, 0, stream>>>(
            x, statm, stati, g, b, add, outf, (unsigned short*)outbf, Mo, C, Cp, mode);
    };

    // swizzled bf16 weights: [ntaps][CIp/32][CO/16][lane][elem]
    auto mkswz = [&](int idx, int nt, int CI, int CO) -> unsigned short* {
        int CIp = pad32(CI);
        size_t n = (size_t)nt * CIp * CO;     // (CIp/32)*(CO/16)*512 per tap
        unsigned short* p = (unsigned short*)alloc(n * 2);
        k_swz<<<nblk(n), 256, 0, stream>>>((const float*)d_in[idx], p,
                                           nt, CI, CIp, CO, n);
        return p;
    };

    // ---- weights ----
    unsigned short* stemW = mkswz(iStemW, 27, 3, 16);
    unsigned short *Scw[4], *Sw1[4], *Sw2[4], *Swd[4];
    unsigned short *Ucw[4], *Uw1[4], *Uw2[4], *Uwd[4];
    for (int s = 0; s < 4; ++s) {
        int ci = CS[s], co = CS[s + 1];
        Scw[s] = mkswz(S[s].cw, 8, ci, ci);
        Sw1[s] = mkswz(S[s].w1, 27, ci, co);
        Sw2[s] = mkswz(S[s].w2, 27, co, co);
        Swd[s] = mkswz(S[s].wd, 1, ci, co);
    }
    for (int u = 0; u < 4; ++u) {
        int ci = CS[4 + u], co = CS[5 + u], cc = co + CS[3 - u];
        Ucw[u] = mkswz(U[u].cw, 8, ci, co);
        Uw1[u] = mkswz(U[u].w1, 27, cc, co);
        Uw2[u] = mkswz(U[u].w2, 27, co, co);
        Uwd[u] = mkswz(U[u].wd, 1, cc, co);
    }
    unsigned short* TW[5];
    int Cq[5] = {CS[8], CS[7], CS[6], CS[5], CS[4]};   // 32,64,128,256,256
    for (int q = 0; q < 5; ++q) TW[q] = mkswz(iTW[q], 1, Cq[q], 32);

    // ---- persistent activations ----
    float* vt3f = (float*)alloc((size_t)M0 * 3 * 4);
    unsigned short* vt3bf = (unsigned short*)alloc((size_t)M0 * 32 * 2);
    float* dsf[5]; unsigned short* dsbf[5]; int dsCp[5];
    dsCp[0] = 32;
    dsf[0]  = (float*)alloc((size_t)M0 * 16 * 4);
    dsbf[0] = (unsigned short*)alloc((size_t)M0 * 32 * 2);
    float* x0f = (float*)alloc((size_t)M0 * 16 * 4);
    unsigned short* x0bf = (unsigned short*)alloc((size_t)M0 * 32 * 2);
    for (int l = 1; l < 5; ++l) {
        dsCp[l] = pad32(CS[l]);
        dsf[l]  = (float*)alloc((size_t)M[l] * CS[l] * 4);
        dsbf[l] = (unsigned short*)alloc((size_t)M[l] * dsCp[l] * 2);
    }
    float* upf[4]; unsigned short* upbf[4];
    for (int u = 0; u < 4; ++u) {
        int lev = 3 - u, co = CS[5 + u];
        upf[u]  = (float*)alloc((size_t)M[lev] * co * 4);
        upbf[u] = (unsigned short*)alloc((size_t)M[lev] * co * 2);
    }
    float* ptf = (float*)alloc((size_t)Npts * 256 * 4);
    unsigned short* ptbf = (unsigned short*)alloc((size_t)Npts * 256 * 2);

    // ---- scratch ----
    float* t1 = (float*)alloc(tmax * 4);
    float* t2 = (float*)alloc(tmax * 4);
    float* t3 = (float*)alloc(tmax * 4);
    unsigned short* h1bf  = (unsigned short*)alloc(tmax * 2);   // always Cp==C
    unsigned short* catbf = (unsigned short*)alloc(catmax * 2);

    auto resblock = [&](const unsigned short* xbf, int Mo, int CI_, int CO_,
                        const int* n27,
                        const void* w1, const float* g1, const float* b1,
                        const void* w2, const float* g2, const float* b2,
                        const void* wd, const float* gd, const float* bd,
                        float* outf, unsigned short* outbf, int outCp) {
        int CIp_ = pad32(CI_);
        conv(xbf, w1, n27, t1, Mo, CIp_, CO_, 27);
        bn(t1, Mo, CO_, CO_, g1, b1, 1, nullptr, nullptr, h1bf); // relu(bn(conv1))
        conv(h1bf, w2, n27, t2, Mo, CO_, CO_, 27);               // conv2
        conv(xbf, wd, nullptr, t3, Mo, CIp_, CO_, 1);            // x @ wd
        bn(t3, Mo, CO_, CO_, gd, bd, 0, nullptr, t3, nullptr);   // bn(sc)
        bn(t2, Mo, CO_, outCp, g2, b2, 2, t3, outf, outbf);      // relu(bn+sc)
    };

    // =========================== stem ===========================
    k_zero_f32<<<nblk((size_t)M0 * 3), 256, 0, stream>>>(vt3f, (size_t)M0 * 3);
    k_scatter_add<<<nblk((size_t)Npts * 3), 256, 0, stream>>>(
        P(0), I(iP2V), vt3f, Npts, 3);
    k_div_counts<<<nblk((size_t)M0 * 32), 256, 0, stream>>>(
        vt3f, P(iCnt), vt3bf, M0, 3, 32);
    conv(vt3bf, stemW, I(iN27[0]), t1, M0, 32, 16, 27);
    bn(t1, M0, 16, 32, P(iStemG), P(iStemB), 1, nullptr, dsf[0], dsbf[0]); // vt
    k_devox<<<nblk((size_t)Npts * 16), 256, 0, stream>>>(
        dsf[0], I(iVI[0]), P(iVW[0]), ptf, nullptr, Npts, 16);             // pt0
    k_zero_f32<<<nblk((size_t)M0 * 16), 256, 0, stream>>>(x0f, (size_t)M0 * 16);
    k_scatter_add<<<nblk((size_t)Npts * 16), 256, 0, stream>>>(
        ptf, I(iP2V), x0f, Npts, 16);
    k_div_counts<<<nblk((size_t)M0 * 32), 256, 0, stream>>>(
        x0f, P(iCnt), x0bf, M0, 16, 32);                                   // x

    // =========================== encoder ===========================
    const unsigned short* prevbf = x0bf;
    for (int s = 0; s < 4; ++s) {
        int ci = CS[s], co = CS[s + 1], Mo = M[s + 1];
        conv(prevbf, Scw[s], I(iDown[s + 1]), t1, Mo, pad32(ci), ci, 8);
        bn(t1, Mo, ci, pad32(ci), P(S[s].cg), P(S[s].cb), 1, nullptr,
           nullptr, catbf);
        resblock(catbf, Mo, ci, co, I(iN27[s + 1]),
                 Sw1[s], P(S[s].g1), P(S[s].b1),
                 Sw2[s], P(S[s].g2), P(S[s].b2),
                 Swd[s], P(S[s].gd), P(S[s].bd),
                 dsf[s + 1], dsbf[s + 1], dsCp[s + 1]);
        prevbf = dsbf[s + 1];
    }

    // =========================== decoder ===========================
    const unsigned short* inbf = dsbf[4];
    int inCp = dsCp[4];
    for (int u = 0; u < 4; ++u) {
        int lev = 3 - u, ci = CS[4 + u], co = CS[5 + u], sk = CS[3 - u];
        int Mo = M[lev], cc = co + sk, ccp = pad32(cc);
        conv(inbf, Ucw[u], I(iDec[4 - u]), t1, Mo, inCp, co, 8);
        bn(t1, Mo, co, co, P(U[u].cg), P(U[u].cb), 1, nullptr, nullptr, h1bf);
        k_copy_cols<<<nblk((size_t)Mo * co), 256, 0, stream>>>(
            h1bf, catbf, Mo, co, co, ccp, 0);
        k_copy_cols<<<nblk((size_t)Mo * sk), 256, 0, stream>>>(
            dsbf[3 - u], catbf, Mo, sk, dsCp[3 - u], ccp, co);  // skips ds3..vt
        if (ccp > cc)
            k_pad_cols<<<nblk((size_t)Mo * (ccp - cc)), 256, 0, stream>>>(
                catbf, Mo, ccp, cc, ccp);
        resblock(catbf, Mo, cc, co, I(iN27[lev]),
                 Uw1[u], P(U[u].g1), P(U[u].b1),
                 Uw2[u], P(U[u].g2), P(U[u].b2),
                 Uwd[u], P(U[u].gd), P(U[u].bd),
                 upf[u], upbf[u], co);
        inbf = upbf[u];
        inCp = co;
    }

    // ====================== multi-scale head ======================
    // out = sum_q relu(bn(devox(fv_q, v2p_q) @ t_{q+1}_w))
    float* fvs[5] = { upf[3], upf[2], upf[1], upf[0], dsf[4] };
    for (int q = 0; q < 5; ++q) {
        int C = Cq[q];                               // 32,64,128,256,256
        k_devox<<<nblk((size_t)Npts * C), 256, 0, stream>>>(
            fvs[q], I(iVI[q]), P(iVW[q]), ptf, ptbf, Npts, C);
        conv(ptbf, TW[q], nullptr, t1, Npts, C, 32, 1);
        bn(t1, Npts, 32, 32, P(iTg[q]), P(iTb[q]),
           q == 0 ? 1 : 3,
           q == 0 ? nullptr : (const float*)d_out,
           (float*)d_out, nullptr);
    }
}
